// MyGnnHardCoded_45956150067830
// MI455X (gfx1250) — matrix operand order repaired and checked by reading, not compiled, not run on hardware
//
#include <hip/hip_runtime.h>
#include <hip/hip_bf16.h>
#include <math.h>

typedef __attribute__((ext_vector_type(16))) _Float16 v16h;
typedef __attribute__((ext_vector_type(8)))  _Float16 v8h;
typedef __attribute__((ext_vector_type(2)))  _Float16 v2h;
typedef __attribute__((ext_vector_type(8)))  float    v8f;

static constexpr int NNODE = 50000;
static constexpr int NE    = 800000;
static constexpr int NE2   = NE + NNODE;   // edges + self loops

// ---------------- helpers ----------------

__device__ __forceinline__ void atomic_max_float(float* addr, float v) {
  // standard IEEE bit trick: works with buffer initialized to -inf
  if (v >= 0.f) atomicMax((int*)addr, __float_as_int(v));
  else          atomicMin((unsigned int*)addr, __float_as_uint(v));
}

// A fragment (16x32 f16): lanes 0-15 -> M=lane, K 0..15; lanes 16-31 -> K 16..31.
// Per lane the 16 halves are two contiguous 8-half runs at +0 and +8 -> 2x ds_load_b128.
__device__ __forceinline__ v16h load_a_frag(const _Float16* sA, int lda, int row0, int kbase, int lane) {
  const int m = lane & 15;
  const int khalf = (lane >> 4) << 4;
  const _Float16* p = sA + (row0 + m) * lda + kbase + khalf;
  v8h lo = *reinterpret_cast<const v8h*>(p);
  v8h hi = *reinterpret_cast<const v8h*>(p + 8);
  return __builtin_shufflevector(lo, hi, 0, 1, 2, 3, 4, 5, 6, 7, 8, 9, 10, 11, 12, 13, 14, 15);
}

// B fragment from *swizzled* LDS: tile-major, per-lane 16 halves contiguous (32B) -> 2x ds_load_b128.
__device__ __forceinline__ v16h load_b_frag_sw(const _Float16* sW, int tile, int lane) {
  return *reinterpret_cast<const v16h*>(sW + ((tile * 32 + lane) << 4));
}

// ---------------- utility kernels ----------------

__global__ void k_fill(float* p, long long n, float v) {
  long long i = (long long)blockIdx.x * blockDim.x + threadIdx.x;
  if (i < n) p[i] = v;
}

__global__ void k_build(const int* __restrict__ ei, int* __restrict__ src2,
                        int* __restrict__ dst2, float* __restrict__ deg) {
  int i = blockIdx.x * blockDim.x + threadIdx.x;
  if (i >= NE2) return;
  int s = (i < NE) ? ei[i]      : (i - NE);
  int d = (i < NE) ? ei[NE + i] : (i - NE);
  src2[i] = s; dst2[i] = d;
  atomicAdd(&deg[d], 1.0f);
}

__global__ void k_dinv(const float* __restrict__ deg, float* __restrict__ dinv) {
  int i = blockIdx.x * blockDim.x + threadIdx.x;
  if (i < NNODE) dinv[i] = rsqrtf(fmaxf(deg[i], 1e-12f));
}

// ---------------- fused PointNet edge MLP (WMMA) ----------------
// per 256-thread block: 128 edges. Gather [x[src], pos[src]-pos[dst]] -> LDS f16,
// GEMM(64x64)+bias+relu -> LDS f16 -> GEMM(64x64)+bias -> atomic segment-max by dst.
__launch_bounds__(256)
__global__ void k_pointnet(const float* __restrict__ x, const float* __restrict__ pos,
                           const int* __restrict__ src2, const int* __restrict__ dst2,
                           const float* __restrict__ w1, const float* __restrict__ b1,
                           const float* __restrict__ w2, const float* __restrict__ b2,
                           float* __restrict__ nodemax)
{
  __shared__ alignas(32) _Float16 sW1[64 * 64];   // swizzled fragment-major
  __shared__ alignas(32) _Float16 sW2[64 * 64];   // swizzled fragment-major
  __shared__ float    sb1[64], sb2[64];
  __shared__ alignas(32) _Float16 sA[128 * 72];   // padded stride (144B = 9*16B)
  __shared__ alignas(32) _Float16 sMid[128 * 72];
  const int tid = threadIdx.x;
  // stage weights swizzled: index = ((kt*4+nt)*32 + lane)*16 + e
  for (int i = tid; i < 4096; i += 256) {
    const int e = i & 15, l = (i >> 4) & 31, tile = i >> 9;
    const int nt = tile & 3, kt = tile >> 2;
    const int k = kt * 32 + ((l >> 4) << 4) + e;
    const int n = nt * 16 + (l & 15);
    sW1[i] = (_Float16)w1[k * 64 + n];
    sW2[i] = (_Float16)w2[k * 64 + n];
  }
  if (tid < 64) { sb1[tid] = b1[tid]; sb2[tid] = b2[tid]; }
  const int e0 = blockIdx.x * 128;
  // gather edge features as f16 pairs
  for (int i = tid; i < 128 * 32; i += 256) {
    const int r = i >> 5, c0 = (i & 31) << 1;
    const int e = e0 + r;
    float v0 = 0.f, v1 = 0.f;
    if (e < NE2) {
      const int s = src2[e];
      const int d = dst2[e];
#pragma unroll
      for (int j = 0; j < 2; ++j) {
        const int c = c0 + j;
        float v;
        if (c < 61) v = x[(size_t)s * 61 + c];
        else        v = pos[s * 3 + (c - 61)] - pos[d * 3 + (c - 61)];
        if (j == 0) v0 = v; else v1 = v;
      }
    }
    v2h pr = { (_Float16)v0, (_Float16)v1 };
    *reinterpret_cast<v2h*>(&sA[r * 72 + c0]) = pr;
  }
  __syncthreads();

  const int lane = tid & 31;
  const int wrow = (tid >> 5) * 16;            // 16-row stripe per wave (wave32)
  v8f zf = {};
  v8f acc[4] = {zf, zf, zf, zf};
#pragma unroll
  for (int kt = 0; kt < 2; ++kt) {
    v16h a = load_a_frag(sA, 72, wrow, kt * 32, lane);
#pragma unroll
    for (int nt = 0; nt < 4; ++nt) {
      v16h b = load_b_frag_sw(sW1, kt * 4 + nt, lane);
      acc[nt] = __builtin_amdgcn_wmma_f32_16x16x32_f16(false, a, false, b, (short)0, acc[nt], false, false);
    }
  }
#pragma unroll
  for (int nt = 0; nt < 4; ++nt) {
    const int col = nt * 16 + (lane & 15);
#pragma unroll
    for (int r = 0; r < 8; ++r) {
      const int row = wrow + ((lane >> 4) << 3) + r;
      float v = fmaxf(acc[nt][r] + sb1[col], 0.f);
      sMid[row * 72 + col] = (_Float16)v;     // intra-wave LDS: in-order, no barrier needed
    }
  }
  v8f acc2[4] = {zf, zf, zf, zf};
#pragma unroll
  for (int kt = 0; kt < 2; ++kt) {
    v16h a = load_a_frag(sMid, 72, wrow, kt * 32, lane);
#pragma unroll
    for (int nt = 0; nt < 4; ++nt) {
      v16h b = load_b_frag_sw(sW2, kt * 4 + nt, lane);
      acc2[nt] = __builtin_amdgcn_wmma_f32_16x16x32_f16(false, a, false, b, (short)0, acc2[nt], false, false);
    }
  }
#pragma unroll
  for (int nt = 0; nt < 4; ++nt) {
    const int col = nt * 16 + (lane & 15);
#pragma unroll
    for (int r = 0; r < 8; ++r) {
      const int row = wrow + ((lane >> 4) << 3) + r;
      const int e = e0 + row;
      if (e < NE2) {
        const float v = acc2[nt][r] + sb2[col];
        atomic_max_float(&nodemax[(size_t)dst2[e] * 64 + col], v);
      }
    }
  }
}

// ---------------- generic node GEMM (WMMA), C = [relu](A[MxK] @ W[KxNF] + b) ----------------
template<int K, int NF, bool RELU, bool BIAS>
__launch_bounds__(256)
__global__ void k_gemm(const float* __restrict__ A, const float* __restrict__ W,
                       const float* __restrict__ bias, float* __restrict__ C, int M)
{
  constexpr int LDA = K + 8;
  constexpr int NT = NF / 16;
  constexpr int KT = K / 32;
  __shared__ alignas(32) _Float16 sW[K * NF];     // swizzled fragment-major
  __shared__ alignas(32) _Float16 sA[128 * LDA];
  __shared__ float    sB[NF];
  const int tid = threadIdx.x;
  // stage W swizzled: index = ((kt*NT+nt)*32 + lane)*16 + e
  for (int i = tid; i < K * NF; i += 256) {
    const int e = i & 15, l = (i >> 4) & 31, tile = i >> 9;
    const int nt = tile % NT, kt = tile / NT;
    const int k = kt * 32 + ((l >> 4) << 4) + e;
    const int n = nt * 16 + (l & 15);
    sW[i] = (_Float16)W[k * NF + n];
  }
  if (BIAS) { for (int i = tid; i < NF; i += 256) sB[i] = bias[i]; }
  const int row0 = blockIdx.x * 128;
  for (int i = tid; i < 128 * K; i += 256) {
    const int r = i / K, c = i % K;       // K is power of 2 -> shifts
    const int gr = row0 + r;
    sA[r * LDA + c] = (gr < M) ? (_Float16)A[(size_t)gr * K + c] : (_Float16)0.f;
  }
  if (row0 + 128 < M)                      // speculative prefetch of next A tile
    __builtin_prefetch(A + (size_t)(row0 + 128) * K + tid * 4, 0, 0);
  __syncthreads();

  const int lane = tid & 31;
  const int wrow = (tid >> 5) * 16;
  v8f zf = {};
  v8f acc[NT];
#pragma unroll
  for (int i = 0; i < NT; ++i) acc[i] = zf;
#pragma unroll
  for (int kt = 0; kt < KT; ++kt) {
    v16h a = load_a_frag(sA, LDA, wrow, kt * 32, lane);
#pragma unroll
    for (int nt = 0; nt < NT; ++nt) {
      v16h b = load_b_frag_sw(sW, kt * NT + nt, lane);
      acc[nt] = __builtin_amdgcn_wmma_f32_16x16x32_f16(false, a, false, b, (short)0, acc[nt], false, false);
    }
  }
#pragma unroll
  for (int nt = 0; nt < NT; ++nt) {
    const int col = nt * 16 + (lane & 15);
    const float bv = BIAS ? sB[col] : 0.f;
#pragma unroll
    for (int r = 0; r < 8; ++r) {
      const int grow = row0 + wrow + ((lane >> 4) << 3) + r;
      if (grow < M) {
        float v = acc[nt][r] + bv;
        if (RELU) v = fmaxf(v, 0.f);
        C[(size_t)grow * NF + col] = v;
      }
    }
  }
}

// ---------------- GAT edge phases ----------------

__global__ void k_scores(const float* __restrict__ xp, const float* __restrict__ asrc,
                         const float* __restrict__ adst, float* __restrict__ ssrc,
                         float* __restrict__ sdst, int fo) {
  int i = blockIdx.x * blockDim.x + threadIdx.x;
  if (i >= NNODE) return;
  float s1 = 0.f, s2 = 0.f;
  for (int f = 0; f < fo; ++f) {
    float v = xp[(size_t)i * fo + f];
    s1 += v * asrc[f]; s2 += v * adst[f];
  }
  ssrc[i] = s1; sdst[i] = s2;
}

__global__ void k_edge_max(const int* __restrict__ src2, const int* __restrict__ dst2,
                           const float* __restrict__ ssrc, const float* __restrict__ sdst,
                           float* __restrict__ escore, float* __restrict__ m) {
  int i = blockIdx.x * blockDim.x + threadIdx.x;
  if (i >= NE2) return;
  float e = ssrc[src2[i]] + sdst[dst2[i]];
  e = (e > 0.f) ? e : 0.2f * e;            // leaky relu 0.2
  escore[i] = e;
  atomic_max_float(&m[dst2[i]], e);
}

__global__ void k_edge_exp(const int* __restrict__ dst2, float* __restrict__ escore,
                           const float* __restrict__ m, float* __restrict__ ssum) {
  int i = blockIdx.x * blockDim.x + threadIdx.x;
  if (i >= NE2) return;
  const int d = dst2[i];
  const float ex = __expf(escore[i] - m[d]);
  escore[i] = ex;
  atomicAdd(&ssum[d], ex);
}

__global__ void k_edge_agg(const int* __restrict__ src2, const int* __restrict__ dst2,
                           const float* __restrict__ escore, const float* __restrict__ ssum,
                           const float* __restrict__ xp, float* __restrict__ agg, int foshift) {
  long long t = (long long)blockIdx.x * blockDim.x + threadIdx.x;
  const long long total = (long long)NE2 << foshift;
  if (t >= total) return;
  const int e = (int)(t >> foshift);
  const int f = (int)(t & ((1 << foshift) - 1));
  const int s = src2[e], d = dst2[e];
  const float alpha = escore[e] / ssum[d];
  atomicAdd(&agg[((size_t)d << foshift) + f], alpha * xp[((size_t)s << foshift) + f]);
}

__global__ void k_bias_relu(const float* __restrict__ agg, const float* __restrict__ bias,
                            float* __restrict__ hout, int fo) {
  long long i = (long long)blockIdx.x * blockDim.x + threadIdx.x;
  if (i >= (long long)NNODE * fo) return;
  const int f = (int)(i & (fo - 1));
  hout[i] = fmaxf(agg[i] + bias[f], 0.f);
}

// ---------------- GCN output ----------------

__global__ void k_gcn_agg(const int* __restrict__ src2, const int* __restrict__ dst2,
                          const float* __restrict__ dinv, const float* __restrict__ xp,
                          float* __restrict__ out) {
  long long t = (long long)blockIdx.x * blockDim.x + threadIdx.x;
  if (t >= (long long)NE2 * 16) return;
  const int e = (int)(t >> 4), f = (int)(t & 15);
  const int s = src2[e], d = dst2[e];
  atomicAdd(&out[(size_t)d * 16 + f], dinv[s] * dinv[d] * xp[(size_t)s * 16 + f]);
}

__global__ void k_gcn_bias(float* __restrict__ out, const float* __restrict__ b) {
  int i = blockIdx.x * blockDim.x + threadIdx.x;
  if (i < NNODE * 16) out[i] += b[i & 15];
}

// ---------------- host ----------------

extern "C" void kernel_launch(void* const* d_in, const int* in_sizes, int n_in,
                              void* d_out, int out_size, void* d_ws, size_t ws_size,
                              hipStream_t stream) {
  (void)in_sizes; (void)n_in; (void)out_size; (void)ws_size;
  const float* x   = (const float*)d_in[0];
  const float* pos = (const float*)d_in[1];
  const int*   ei  = (const int*)d_in[2];
  const float* w1  = (const float*)d_in[3];
  const float* b1  = (const float*)d_in[4];
  const float* w2  = (const float*)d_in[5];
  const float* b2  = (const float*)d_in[6];
  const float* g1w = (const float*)d_in[7];
  const float* g1b = (const float*)d_in[8];
  const float* g2w = (const float*)d_in[9];
  const float* g2b = (const float*)d_in[10];
  const float* g3w = (const float*)d_in[11];
  const float* g3b = (const float*)d_in[12];
  const float* gcw = (const float*)d_in[33];
  const float* gcb = (const float*)d_in[34];
  float* out = (float*)d_out;

  char* ws = (char*)d_ws;
  size_t off = 0;
  auto take = [&](size_t bytes) { char* p = ws + off; off = (off + bytes + 255) & ~(size_t)255; return p; };
  int*   src2   = (int*)  take((size_t)NE2 * 4);
  int*   dst2   = (int*)  take((size_t)NE2 * 4);
  float* deg    = (float*)take((size_t)NNODE * 4);
  float* dinv   = (float*)take((size_t)NNODE * 4);
  float* m      = (float*)take((size_t)NNODE * 4);
  float* ssum   = (float*)take((size_t)NNODE * 4);
  float* ssrc   = (float*)take((size_t)NNODE * 4);
  float* sdst   = (float*)take((size_t)NNODE * 4);
  float* escore = (float*)take((size_t)NE2 * 4);
  float* hA     = (float*)take((size_t)NNODE * 128 * 4);
  float* hB     = (float*)take((size_t)NNODE * 128 * 4);
  float* hC     = (float*)take((size_t)NNODE * 128 * 4);

  const int T = 256;
  auto nb = [](long long n) { return (unsigned)((n + 255) / 256); };
  const unsigned gtiles = (NNODE + 127) / 128;

  // build edges + degree
  k_fill<<<nb(NNODE), T, 0, stream>>>(deg, NNODE, 0.f);
  k_build<<<nb(NE2), T, 0, stream>>>(ei, src2, dst2, deg);

  // PointNetConv: fused edge MLP + segment max into hA[N,64]
  k_fill<<<nb((long long)NNODE * 64), T, 0, stream>>>(hA, (long long)NNODE * 64, -INFINITY);
  k_pointnet<<<(NE2 + 127) / 128, T, 0, stream>>>(x, pos, src2, dst2, w1, b1, w2, b2, hA);

  // node MLP g1,g2,g3
  k_gemm<64, 32, true,  true><<<gtiles, T, 0, stream>>>(hA, g1w, g1b, hB, NNODE);
  k_gemm<32, 128, true, true><<<gtiles, T, 0, stream>>>(hB, g2w, g2b, hC, NNODE);
  k_gemm<128, 64, false, true><<<gtiles, T, 0, stream>>>(hC, g3w, g3b, hB, NNODE);

  // 5x GAT layers: h in hB, xp in hA, agg in hC, new h -> hB
#define GAT_LAYER(IDX, FI, FO, FOSHIFT)                                                              \
  do {                                                                                               \
    const float* gw  = (const float*)d_in[13 + 4 * (IDX)];                                           \
    const float* gas = (const float*)d_in[14 + 4 * (IDX)];                                           \
    const float* gad = (const float*)d_in[15 + 4 * (IDX)];                                           \
    const float* gb  = (const float*)d_in[16 + 4 * (IDX)];                                           \
    k_gemm<FI, FO, false, false><<<gtiles, T, 0, stream>>>(hB, gw, nullptr, hA, NNODE);              \
    k_scores<<<nb(NNODE), T, 0, stream>>>(hA, gas, gad, ssrc, sdst, FO);                             \
    k_fill<<<nb(NNODE), T, 0, stream>>>(m, NNODE, -INFINITY);                                        \
    k_fill<<<nb(NNODE), T, 0, stream>>>(ssum, NNODE, 0.f);                                           \
    k_fill<<<nb((long long)NNODE * FO), T, 0, stream>>>(hC, (long long)NNODE * FO, 0.f);             \
    k_edge_max<<<nb(NE2), T, 0, stream>>>(src2, dst2, ssrc, sdst, escore, m);                        \
    k_edge_exp<<<nb(NE2), T, 0, stream>>>(dst2, escore, m, ssum);                                    \
    k_edge_agg<<<nb((long long)NE2 << (FOSHIFT)), T, 0, stream>>>(src2, dst2, escore, ssum, hA, hC,  \
                                                                  FOSHIFT);                          \
    k_bias_relu<<<nb((long long)NNODE * FO), T, 0, stream>>>(hC, gb, hB, FO);                        \
  } while (0)

  GAT_LAYER(0, 64, 128, 7);
  GAT_LAYER(1, 128, 32, 5);
  GAT_LAYER(2, 32, 128, 7);
  GAT_LAYER(3, 128, 64, 6);
  GAT_LAYER(4, 64, 64, 6);
#undef GAT_LAYER

  // GCN output
  k_dinv<<<nb(NNODE), T, 0, stream>>>(deg, dinv);
  k_gemm<64, 16, false, false><<<gtiles, T, 0, stream>>>(hB, gcw, nullptr, hA, NNODE);
  k_fill<<<nb((long long)NNODE * 16), T, 0, stream>>>(out, (long long)NNODE * 16, 0.f);
  k_gcn_agg<<<nb((long long)NE2 * 16), T, 0, stream>>>(src2, dst2, dinv, hA, out);
  k_gcn_bias<<<nb((long long)NNODE * 16), T, 0, stream>>>(out, gcb);
}
